// BidafAttn_20203526161039
// MI455X (gfx1250) — matrix operand order repaired and checked by reading, not compiled, not run on hardware
//
#include <hip/hip_runtime.h>
#include <hip/hip_bf16.h>

typedef __attribute__((ext_vector_type(16))) __bf16 v16bf;
typedef __attribute__((ext_vector_type(8)))  __bf16 v8bf;
typedef __attribute__((ext_vector_type(8)))  float  v8f;

#define B_   32
#define T1   1024
#define T2   1024
#define D_   256
#define NEG_BIG (-1e30f)

union V16u { v16bf v; v8bf h[2]; };
static __device__ __forceinline__ v16bf pack2(v8bf lo, v8bf hi) {
    V16u u; u.h[0] = lo; u.h[1] = hi; return u.v;
}

// CDNA5 async global->LDS copy (ASYNCcnt-tracked, no VGPR data path).
// GVS mode: saddr = uniform 64-bit base (SGPR pair), vaddr = per-lane 32-bit
// byte offset, vdst = LDS byte address (low 32 bits of the generic pointer).
static __device__ __forceinline__ void async_b128(unsigned lds_addr, unsigned voff,
                                                  const void* base) {
    asm volatile("global_load_async_to_lds_b128 %0, %1, %2"
                 :: "v"(lds_addr), "v"(voff), "s"(base)
                 : "memory");
}
static __device__ __forceinline__ void wait_async0() {
    asm volatile("s_wait_asynccnt 0" ::: "memory");
}

// ---- pack Q = bf16(s1 * w3) ------------------------------------------------
__global__ void qpack_kernel(const float* __restrict__ s1, const float* __restrict__ w,
                             __bf16* __restrict__ Q) {
    int r = blockIdx.x;      // b*T1 + t
    int d = threadIdx.x;     // 0..255
    float q = s1[(size_t)r * D_ + d] * w[2 * D_ + d];
    Q[(size_t)r * D_ + d] = (__bf16)q;
}

// ---- pack K = bf16(s2), Vt = bf16(s2^T), biasm = mask ? s2.w2 : -1e30 ------
__global__ void kvpack_kernel(const float* __restrict__ s2, const float* __restrict__ w,
                              const int* __restrict__ l2,
                              __bf16* __restrict__ K, __bf16* __restrict__ Vt,
                              float* __restrict__ biasm) {
    int r = blockIdx.x;      // b*T2 + s
    int d = threadIdx.x;     // 0..255
    int b = r >> 10;         // T2 == 1024
    int srow = r & (T2 - 1);
    float v = s2[(size_t)r * D_ + d];
    K[(size_t)r * D_ + d] = (__bf16)v;
    Vt[((size_t)b * D_ + d) * T2 + srow] = (__bf16)v;

    __shared__ float red[256];
    red[d] = v * w[D_ + d];
    __syncthreads();
    for (int s = 128; s > 0; s >>= 1) {
        if (d < s) red[d] += red[d + s];
        __syncthreads();
    }
    if (d == 0) biasm[r] = (srow < l2[b]) ? red[0] : NEG_BIG;
}

// issue async staging of one K/V chunk (32 t2-columns): 16KB K + 16KB Vt
static __device__ __forceinline__ void stage_chunk(const __bf16* Kb, const __bf16* Vb,
                                                   int j, unsigned ldsK, unsigned ldsV,
                                                   int tid) {
    const void* kbase = (const void*)(Kb + (size_t)j * 32 * D_);   // contiguous 16KB
    const void* vbase = (const void*)(Vb + (size_t)j * 32);        // 256 rows x 64B
    #pragma unroll
    for (int i = 0; i < 8; i++) {
        unsigned f = (unsigned)(tid + i * 128);                    // 0..1023 x16B
        async_b128(ldsK + f * 16u, f * 16u, kbase);
    }
    #pragma unroll
    for (int i = 0; i < 8; i++) {
        unsigned f = (unsigned)(tid + i * 128);                    // 0..1023
        unsigned goff = (f >> 2) * (unsigned)(T2 * 2) + (f & 3u) * 16u;
        async_b128(ldsV + f * 16u, goff, vbase);
    }
}

// ---- flash attention: per wave a 16-row t1 tile, K/V double-buffered in LDS
__launch_bounds__(128, 1)
__global__ void attn_kernel(const __bf16* __restrict__ Q, const __bf16* __restrict__ K,
                            const __bf16* __restrict__ Vt, const float* __restrict__ biasm,
                            const int* __restrict__ l1, const int* __restrict__ l2,
                            float* __restrict__ out) {
    const int b    = blockIdx.y;
    const int wv   = threadIdx.y;            // wave id 0..3
    const int lane = threadIdx.x;            // 0..31
    const int m    = lane & 15;
    const int hi   = lane >> 4;
    const int rowbase = blockIdx.x * 64 + wv * 16;
    const int tid  = wv * 32 + lane;         // 0..127

    __shared__ __bf16 sQ[4][16 * D_];        // 32KB: per-wave Q tiles
    __shared__ __bf16 sK[2][32 * D_];        // 2 x 16KB: K chunk double buffer
    __shared__ __bf16 sV[2][D_ * 32];        // 2 x 16KB: Vt chunk double buffer
    __shared__ __bf16 sP[4][16 * 32];        // per-wave prob relayout

    const unsigned ldsQ   = (unsigned)(uintptr_t)&sQ[wv][0];
    const unsigned ldsKb0 = (unsigned)(uintptr_t)&sK[0][0];
    const unsigned ldsKb1 = (unsigned)(uintptr_t)&sK[1][0];
    const unsigned ldsVb0 = (unsigned)(uintptr_t)&sV[0][0];
    const unsigned ldsVb1 = (unsigned)(uintptr_t)&sV[1][0];

    // async-stage this wave's Q tile (16 rows x 256 bf16 = 512 x 16B)
    {
        const void* Qbase = (const void*)(Q + ((size_t)b * T1 + rowbase) * D_);
        #pragma unroll
        for (int i = 0; i < 16; i++) {
            unsigned f = (unsigned)(lane + i * 32);
            async_b128(ldsQ + f * 16u, f * 16u, Qbase);
        }
    }

    const int l1v = l1[b];
    const int l2v = l2[b];
    int jmax = (l2v + 31) >> 5;
    if (jmax > T2 / 32) jmax = T2 / 32;

    const __bf16* Kb = K  + (size_t)b * T2 * D_;
    const __bf16* Vb = Vt + (size_t)b * D_ * T2;

    v8f O[16] = {};
    float mrow[8], lrow[8];
    #pragma unroll
    for (int r = 0; r < 8; r++) { mrow[r] = NEG_BIG; lrow[r] = 0.f; }

    if (jmax > 0) stage_chunk(Kb, Vb, 0, ldsKb0, ldsVb0, tid);

    for (int j = 0; j < jmax; j++) {
        const int cur = j & 1;
        wait_async0();        // this wave's async writes (chunk j + Q) landed
        __syncthreads();      // all waves wrote buf[cur]; all done reading buf[cur^1]
        if (j + 1 < jmax)     // prefetch next chunk; overlaps the WMMAs below
            stage_chunk(Kb, Vb, j + 1, cur ? ldsKb0 : ldsKb1, cur ? ldsVb0 : ldsVb1, tid);

        // per-lane column bias (l2 mask folded in)
        float bias0 = biasm[(size_t)b * T2 + j * 32 + m];
        float bias1 = biasm[(size_t)b * T2 + j * 32 + 16 + m];

        // S(16x32) = Q(16x256) @ K_chunk^T : two 16x16 subtiles, K-loop over D
        v8f S0 = {}, S1 = {};
        #pragma unroll
        for (int c = 0; c < 8; c++) {
            v16bf A  = pack2(*(const v8bf*)&sQ[wv][m * D_ + c * 32 + hi * 16],
                             *(const v8bf*)&sQ[wv][m * D_ + c * 32 + hi * 16 + 8]);
            v16bf B0 = pack2(*(const v8bf*)&sK[cur][m * D_ + c * 32 + hi * 16],
                             *(const v8bf*)&sK[cur][m * D_ + c * 32 + hi * 16 + 8]);
            v16bf B1 = pack2(*(const v8bf*)&sK[cur][(m + 16) * D_ + c * 32 + hi * 16],
                             *(const v8bf*)&sK[cur][(m + 16) * D_ + c * 32 + hi * 16 + 8]);
            S0 = __builtin_amdgcn_wmma_f32_16x16x32_bf16(false, A, false, B0, (short)0, S0, false, false);
            S1 = __builtin_amdgcn_wmma_f32_16x16x32_bf16(false, A, false, B1, (short)0, S1, false, false);
        }

        // online softmax (row = 8*hi + r lives in VGPR index r; cols across 16 lanes)
        float scale[8], p0[8], p1[8];
        #pragma unroll
        for (int r = 0; r < 8; r++) {
            float x0 = S0[r] + bias0;
            float x1 = S1[r] + bias1;
            float cm = fmaxf(x0, x1);
            cm = fmaxf(cm, __shfl_xor(cm, 1));
            cm = fmaxf(cm, __shfl_xor(cm, 2));
            cm = fmaxf(cm, __shfl_xor(cm, 4));
            cm = fmaxf(cm, __shfl_xor(cm, 8));
            float mn = fmaxf(mrow[r], cm);
            float sc = __expf(mrow[r] - mn);
            float e0 = __expf(x0 - mn);
            float e1 = __expf(x1 - mn);
            float rs = e0 + e1;
            rs += __shfl_xor(rs, 1);
            rs += __shfl_xor(rs, 2);
            rs += __shfl_xor(rs, 4);
            rs += __shfl_xor(rs, 8);
            mrow[r] = mn;
            lrow[r] = lrow[r] * sc + rs;
            scale[r] = sc;
            p0[r] = e0; p1[r] = e1;
        }

        // relayout P through LDS into A-operand order (same-wave DS is in-order)
        #pragma unroll
        for (int r = 0; r < 8; r++) {
            sP[wv][(hi * 8 + r) * 32 + m]      = (__bf16)p0[r];
            sP[wv][(hi * 8 + r) * 32 + 16 + m] = (__bf16)p1[r];
        }
        // rescale accumulator
        #pragma unroll
        for (int t = 0; t < 16; t++) {
            #pragma unroll
            for (int r = 0; r < 8; r++) O[t][r] *= scale[r];
        }
        // O(16x256) += P(16x32) @ V_chunk(32x256), 16 d-tiles
        v16bf Ap = pack2(*(const v8bf*)&sP[wv][m * 32 + hi * 16],
                         *(const v8bf*)&sP[wv][m * 32 + hi * 16 + 8]);
        #pragma unroll
        for (int t = 0; t < 16; t++) {
            v16bf Bv = pack2(*(const v8bf*)&sV[cur][(t * 16 + m) * 32 + hi * 16],
                             *(const v8bf*)&sV[cur][(t * 16 + m) * 32 + hi * 16 + 8]);
            O[t] = __builtin_amdgcn_wmma_f32_16x16x32_bf16(false, Ap, false, Bv, (short)0, O[t], false, false);
        }
        // no trailing barrier: next iteration's wait+barrier covers the hazard
    }

    // normalize + mask rows (t >= l1, or no valid columns) and store
    #pragma unroll
    for (int r = 0; r < 8; r++) {
        int rowlocal = rowbase + hi * 8 + r;
        float inv = (rowlocal < l1v && lrow[r] > 0.f) ? (1.0f / lrow[r]) : 0.f;
        #pragma unroll
        for (int t = 0; t < 16; t++) {
            out[((size_t)b * T1 + rowlocal) * D_ + t * 16 + m] = O[t][r] * inv;
        }
    }
}

extern "C" void kernel_launch(void* const* d_in, const int* in_sizes, int n_in,
                              void* d_out, int out_size, void* d_ws, size_t ws_size,
                              hipStream_t stream) {
    const float* s1 = (const float*)d_in[0];
    const float* s2 = (const float*)d_in[1];
    const float* w  = (const float*)d_in[2];
    const int*   l1 = (const int*)d_in[3];
    const int*   l2 = (const int*)d_in[4];
    float* out = (float*)d_out;

    char* ws = (char*)d_ws;
    size_t qBytes  = (size_t)B_ * T1 * D_ * sizeof(__bf16);
    size_t kBytes  = (size_t)B_ * T2 * D_ * sizeof(__bf16);
    size_t vtBytes = (size_t)B_ * D_ * T2 * sizeof(__bf16);
    __bf16* Q     = (__bf16*)ws;
    __bf16* K     = (__bf16*)(ws + qBytes);
    __bf16* Vt    = (__bf16*)(ws + qBytes + kBytes);
    float*  biasm = (float*)(ws + qBytes + kBytes + vtBytes);

    qpack_kernel<<<dim3(B_ * T1), dim3(256), 0, stream>>>(s1, w, Q);
    kvpack_kernel<<<dim3(B_ * T2), dim3(256), 0, stream>>>(s2, w, l2, K, Vt, biasm);
    attn_kernel<<<dim3(T1 / 64, B_), dim3(32, 4), 0, stream>>>(Q, K, Vt, biasm, l1, l2, out);
}
